// TemporalGNN_78554951843862
// MI455X (gfx1250) — matrix operand order repaired and checked by reading, not compiled, not run on hardware
//
#include <hip/hip_runtime.h>
#include <hip/hip_bf16.h>

// ---------------------------------------------------------------------------
// TemporalGNN on MI455X (gfx1250): DirGCN x2 -> LSTM(8 steps) -> projection.
//  - All dense layers on v_wmma_f32_16x16x32_bf16 (fp32 accumulate).
//  - Weights pre-converted once to bf16 in WMMA B-fragment-friendly layout.
//  - LSTM gate GEMM stages activations block-cooperatively into LDS with
//    global_load_async_to_lds_b128 (ASYNCcnt) and reads A-frags via ds loads.
//  - Graph aggregation on fp32 global atomics (segment sums stay fp32).
// ---------------------------------------------------------------------------

#define GB 4        // batches
#define GS 8        // timesteps
#define GG 32       // GB*GS graphs
#define GN 4000     // nodes per graph
#define GE 32000    // edges per graph
#define GF 64       // input features
#define GH 128      // hidden
#define BN 16000    // GB*GN lstm rows

typedef __attribute__((ext_vector_type(16))) __bf16 v16bf;
typedef __attribute__((ext_vector_type(8)))  float  v8f;

// ---------------- A fragment: fp32 memory/LDS -> bf16 frag -------------------
// A (16x32, MxK): lanes 0-15 row=lane, K 0..15; lanes 16-31 row=lane-16,
// K 16..31.  rowp points at A[row*lda + kb].
__device__ __forceinline__ v16bf load_frag_a(const float* rowp, int hf) {
  const int b0 = hf * 16;
  float4 r0 = *reinterpret_cast<const float4*>(rowp + b0);
  float4 r1 = *reinterpret_cast<const float4*>(rowp + b0 + 4);
  float4 r2 = *reinterpret_cast<const float4*>(rowp + b0 + 8);
  float4 r3 = *reinterpret_cast<const float4*>(rowp + b0 + 12);
  v16bf a;
  a[0]=(__bf16)r0.x;  a[1]=(__bf16)r0.y;  a[2]=(__bf16)r0.z;  a[3]=(__bf16)r0.w;
  a[4]=(__bf16)r1.x;  a[5]=(__bf16)r1.y;  a[6]=(__bf16)r1.z;  a[7]=(__bf16)r1.w;
  a[8]=(__bf16)r2.x;  a[9]=(__bf16)r2.y;  a[10]=(__bf16)r2.z; a[11]=(__bf16)r2.w;
  a[12]=(__bf16)r3.x; a[13]=(__bf16)r3.y; a[14]=(__bf16)r3.z; a[15]=(__bf16)r3.w;
  return a;
}

// B (32x16, KxN): lanes 0-15 col=lane, K 0..15; lanes 16-31 col=lane-16,
// K 16..31.  With W pre-converted to bf16 row-major [Ncols, K], the fragment is
// 16 contiguous bf16 starting at W[col*K + kb + 16*hf] -> direct 32B load.
__device__ __forceinline__ v16bf load_frag_b16(const __bf16* colp, int hf) {
  return *reinterpret_cast<const v16bf*>(colp + 16 * hf);
}

__device__ __forceinline__ float sigf(float x) { return 1.0f / (1.0f + expf(-x)); }

// ---------------- weight f32 -> bf16 ----------------------------------------
__global__ __launch_bounds__(256) void k_cvt_bf16(const float* __restrict__ src,
                                                  __bf16* __restrict__ dst, int n) {
  int i = blockIdx.x * blockDim.x + threadIdx.x;
  if (i < n) dst[i] = (__bf16)src[i];
}

// ---------------- graph normalization ---------------------------------------
__global__ __launch_bounds__(256) void k_degrees(const int* __restrict__ ei,
                                                 const float* __restrict__ ew,
                                                 float* __restrict__ degO,
                                                 float* __restrict__ degI) {
  int idx = blockIdx.x * blockDim.x + threadIdx.x;
  if (idx >= GG * GE) return;
  int g = idx / GE, e = idx - g * GE;
  const int* eg = ei + (size_t)g * 2 * GE;
  int s = eg[e], d = eg[GE + e];
  float w = ew[idx];
  atomicAdd(&degO[g * GN + s], w);
  atomicAdd(&degI[g * GN + d], w);
}

__global__ __launch_bounds__(256) void k_invsqrt(float* __restrict__ deg, int n) {
  int i = blockIdx.x * blockDim.x + threadIdx.x;
  if (i >= n) return;
  float d = deg[i];
  deg[i] = (d > 0.0f) ? rsqrtf(fmaxf(d, 1e-12f)) : 0.0f;
}

__global__ __launch_bounds__(256) void k_norm(const int* __restrict__ ei,
                                              const float* __restrict__ ew,
                                              const float* __restrict__ invO,
                                              const float* __restrict__ invI,
                                              float* __restrict__ nrm) {
  int idx = blockIdx.x * blockDim.x + threadIdx.x;
  if (idx >= GG * GE) return;
  int g = idx / GE, e = idx - g * GE;
  const int* eg = ei + (size_t)g * 2 * GE;
  int s = eg[e], d = eg[GE + e];
  nrm[idx] = ew[idx] * invO[g * GN + s] * invI[g * GN + d];
}

// ---------------- feature scatter (A_hat x, A_hat^T x) -----------------------
__global__ __launch_bounds__(256) void k_scatter(const int* __restrict__ ei,
                                                 const float* __restrict__ nrm,
                                                 const float* __restrict__ x,
                                                 float* __restrict__ aggF,
                                                 float* __restrict__ aggB,
                                                 int Fin) {
  const int chunks = Fin >> 2;
  int idx = blockIdx.x * blockDim.x + threadIdx.x;
  int tot = GG * GE * chunks;
  if (idx >= tot) return;
  int c  = idx % chunks;
  int ge = idx / chunks;
  int e  = ge % GE;
  int g  = ge / GE;
  const int* eg = ei + (size_t)g * 2 * GE;
  int s = eg[e], d = eg[GE + e];
  float w = nrm[(size_t)g * GE + e];
  size_t rs = ((size_t)g * GN + s) * Fin + c * 4;
  size_t rd = ((size_t)g * GN + d) * Fin + c * 4;
  float4 xs = *reinterpret_cast<const float4*>(x + rs);
  float4 xd = *reinterpret_cast<const float4*>(x + rd);
  float* pf = aggF + rd;                 // forward: sum into dst
  atomicAdd(pf + 0, w * xs.x); atomicAdd(pf + 1, w * xs.y);
  atomicAdd(pf + 2, w * xs.z); atomicAdd(pf + 3, w * xs.w);
  float* pb = aggB + rs;                 // backward: sum into src
  atomicAdd(pb + 0, w * xd.x); atomicAdd(pb + 1, w * xd.y);
  atomicAdd(pb + 2, w * xd.z); atomicAdd(pb + 3, w * xd.w);
}

// ---------------- DirGCN projection ------------------------------------------
// One wave per 16-row strip: A fragments held in registers, loop over 8 column
// tiles.  out = 0.5*(aggF Ws^T + bs) + 0.5*(aggB Wd^T + bd)
template <int K>
__global__ __launch_bounds__(256) void k_gcn_gemm(const float* __restrict__ aggF,
                                                  const float* __restrict__ aggB,
                                                  const __bf16* __restrict__ Ws,
                                                  const __bf16* __restrict__ Wd,
                                                  const float* __restrict__ bs,
                                                  const float* __restrict__ bd,
                                                  float* __restrict__ out, int M) {
  const int wid  = blockIdx.x * (blockDim.x >> 5) + (threadIdx.x >> 5);
  const int lane = threadIdx.x & 31;
  if (wid * 16 >= M) return;
  const int hf = lane >> 4, l = lane & 15;
  const int m0 = wid * 16;
  constexpr int KB = K / 32;
  v16bf aF[KB], aB[KB];
  const float* arF = aggF + (size_t)(m0 + l) * K;
  const float* arB = aggB + (size_t)(m0 + l) * K;
#pragma unroll
  for (int k = 0; k < KB; ++k) {
    aF[k] = load_frag_a(arF + 32 * k, hf);
    aB[k] = load_frag_a(arB + 32 * k, hf);
  }
#pragma unroll
  for (int nt = 0; nt < GH / 16; ++nt) {
    const int col = nt * 16 + l;
    const __bf16* wS = Ws + (size_t)col * K;
    const __bf16* wD = Wd + (size_t)col * K;
    v8f accF = {}, accB = {};
#pragma unroll
    for (int k = 0; k < KB; ++k) {
      accF = __builtin_amdgcn_wmma_f32_16x16x32_bf16(false, aF[k], false,
               load_frag_b16(wS + 32 * k, hf), (short)0, accF, false, false);
      accB = __builtin_amdgcn_wmma_f32_16x16x32_bf16(false, aB[k], false,
               load_frag_b16(wD + 32 * k, hf), (short)0, accB, false, false);
    }
    const float bias = 0.5f * (bs[col] + bd[col]);
    float* orow = out + (size_t)(m0 + hf * 8) * GH + col;
#pragma unroll
    for (int r = 0; r < 8; ++r)
      orow[(size_t)r * GH] = 0.5f * (accF[r] + accB[r]) + bias;
  }
}

// ---------------- LSTM gate GEMM ---------------------------------------------
// Block = one 16-row m-tile.  x_t and h rows staged into LDS with async copies,
// A-frags built once per wave from LDS, each wave covers 4 of 32 column tiles.
__global__ __launch_bounds__(256) void k_lstm_gates(const float* __restrict__ seq,
                                                    const float* __restrict__ hstate,
                                                    const __bf16* __restrict__ wih,
                                                    const __bf16* __restrict__ whh,
                                                    const float* __restrict__ bih,
                                                    const float* __restrict__ bhh,
                                                    float* __restrict__ gates, int t) {
  __shared__ float sX[16][132];   // +4 pad: break 512B bank stride
  __shared__ float sH[16][132];
  const int tid = threadIdx.x;
  const int m0  = blockIdx.x * 16;

  // ---- cooperative async stage: 16 rows x 128 floats for x_t and h ----
  {
    const int r = tid >> 4;              // row 0..15
    const int c = (tid & 15) * 4;        // float col {0..60}, plus +64 chunk
    const float* gx = seq    + ((size_t)(m0 + r) * GS + t) * GH;
    const float* gh = hstate + (size_t)(m0 + r) * GH;
    uint32_t dx0 = (uint32_t)(uintptr_t)&sX[r][c];
    uint32_t dx1 = (uint32_t)(uintptr_t)&sX[r][c + 64];
    uint32_t dh0 = (uint32_t)(uintptr_t)&sH[r][c];
    uint32_t dh1 = (uint32_t)(uintptr_t)&sH[r][c + 64];
    asm volatile("global_load_async_to_lds_b128 %0, %1, off"
                 :: "v"(dx0), "v"((uint64_t)(uintptr_t)(gx + c)) : "memory");
    asm volatile("global_load_async_to_lds_b128 %0, %1, off"
                 :: "v"(dx1), "v"((uint64_t)(uintptr_t)(gx + c + 64)) : "memory");
    asm volatile("global_load_async_to_lds_b128 %0, %1, off"
                 :: "v"(dh0), "v"((uint64_t)(uintptr_t)(gh + c)) : "memory");
    asm volatile("global_load_async_to_lds_b128 %0, %1, off"
                 :: "v"(dh1), "v"((uint64_t)(uintptr_t)(gh + c + 64)) : "memory");
    asm volatile("s_wait_asynccnt 0x0" ::: "memory");
  }
  __syncthreads();

  const int lane = tid & 31;
  const int wv   = tid >> 5;             // wave 0..7
  const int hf = lane >> 4, l = lane & 15;

  v16bf ax[4], ah[4];
#pragma unroll
  for (int k = 0; k < 4; ++k) {
    ax[k] = load_frag_a(&sX[l][32 * k], hf);
    ah[k] = load_frag_a(&sH[l][32 * k], hf);
  }

#pragma unroll
  for (int nt = 0; nt < 4; ++nt) {
    const int col = (wv * 4 + nt) * 16 + l;
    const __bf16* w1 = wih + (size_t)col * GH;
    const __bf16* w2 = whh + (size_t)col * GH;
    v8f acc = {};
#pragma unroll
    for (int k = 0; k < 4; ++k) {
      acc = __builtin_amdgcn_wmma_f32_16x16x32_bf16(false, ax[k], false,
              load_frag_b16(w1 + 32 * k, hf), (short)0, acc, false, false);
      acc = __builtin_amdgcn_wmma_f32_16x16x32_bf16(false, ah[k], false,
              load_frag_b16(w2 + 32 * k, hf), (short)0, acc, false, false);
    }
    const float bias = bih[col] + bhh[col];
    float* g = gates + (size_t)(m0 + hf * 8) * (4 * GH) + col;
#pragma unroll
    for (int r = 0; r < 8; ++r)
      g[(size_t)r * (4 * GH)] = acc[r] + bias;
  }
}

// ---------------- LSTM cell update ------------------------------------------
__global__ __launch_bounds__(256) void k_lstm_update(const float* __restrict__ gates,
                                                     float* __restrict__ cstate,
                                                     float* __restrict__ hstate) {
  int idx = blockIdx.x * blockDim.x + threadIdx.x;
  if (idx >= BN * GH) return;
  int p = idx / GH, c = idx - p * GH;
  const float* g = gates + (size_t)p * (4 * GH);
  float ig = g[c], fg = g[GH + c], gg = g[2 * GH + c], og = g[3 * GH + c];
  float cc = sigf(fg) * cstate[idx] + sigf(ig) * tanhf(gg);
  cstate[idx] = cc;
  hstate[idx] = sigf(og) * tanhf(cc);
}

// ---------------- final projection: out = h_last Wp^T + bp -------------------
__global__ __launch_bounds__(256) void k_proj(const float* __restrict__ hstate,
                                              const __bf16* __restrict__ Wp,
                                              const float* __restrict__ bp,
                                              float* __restrict__ out) {
  const int wid  = blockIdx.x * (blockDim.x >> 5) + (threadIdx.x >> 5);
  const int lane = threadIdx.x & 31;
  if (wid * 16 >= BN) return;
  const int hf = lane >> 4, l = lane & 15;
  const int m0 = wid * 16;
  v16bf aR[4];
  const float* ar = hstate + (size_t)(m0 + l) * GH;
#pragma unroll
  for (int k = 0; k < 4; ++k) aR[k] = load_frag_a(ar + 32 * k, hf);
#pragma unroll
  for (int nt = 0; nt < GF / 16; ++nt) {
    const int col = nt * 16 + l;
    const __bf16* wc = Wp + (size_t)col * GH;
    v8f acc = {};
#pragma unroll
    for (int k = 0; k < 4; ++k)
      acc = __builtin_amdgcn_wmma_f32_16x16x32_bf16(false, aR[k], false,
              load_frag_b16(wc + 32 * k, hf), (short)0, acc, false, false);
    const float bias = bp[col];
    float* orow = out + (size_t)(m0 + hf * 8) * GF + col;
#pragma unroll
    for (int r = 0; r < 8; ++r)
      orow[(size_t)r * GF] = acc[r] + bias;
  }
}

// ---------------------------------------------------------------------------
extern "C" void kernel_launch(void* const* d_in, const int* in_sizes, int n_in,
                              void* d_out, int out_size, void* d_ws, size_t ws_size,
                              hipStream_t stream) {
  const float* x_seq = (const float*)d_in[0];
  const int*   ei    = (const int*)  d_in[1];
  const float* ew    = (const float*)d_in[2];
  const float* W0_s  = (const float*)d_in[3];
  const float* b0_s  = (const float*)d_in[4];
  const float* W0_d  = (const float*)d_in[5];
  const float* b0_d  = (const float*)d_in[6];
  const float* W1_s  = (const float*)d_in[7];
  const float* b1_s  = (const float*)d_in[8];
  const float* W1_d  = (const float*)d_in[9];
  const float* b1_d  = (const float*)d_in[10];
  const float* Wih   = (const float*)d_in[11];
  const float* Whh   = (const float*)d_in[12];
  const float* bih   = (const float*)d_in[13];
  const float* bhh   = (const float*)d_in[14];
  const float* Wp    = (const float*)d_in[15];
  const float* bp    = (const float*)d_in[16];
  float* out = (float*)d_out;
  (void)in_sizes; (void)n_in; (void)out_size; (void)ws_size;

  // ---- workspace carve (256B-aligned regions) ----
  char* cur = (char*)d_ws;
  auto carve = [&](size_t bytes) {
    char* p = cur; cur += (bytes + 255) & ~(size_t)255; return p;
  };
  float*  degO  = (float*)carve((size_t)GG * GN * 4);
  float*  degI  = (float*)carve((size_t)GG * GN * 4);
  float*  nrm   = (float*)carve((size_t)GG * GE * 4);
  float*  aggF  = (float*)carve((size_t)GG * GN * GH * 4);
  float*  aggB  = (float*)carve((size_t)GG * GN * GH * 4);
  float*  hbuf  = (float*)carve((size_t)GG * GN * GH * 4);
  float*  gates = (float*)carve((size_t)BN * 4 * GH * 4);
  float*  hst   = (float*)carve((size_t)BN * GH * 4);
  float*  cst   = (float*)carve((size_t)BN * GH * 4);
  __bf16* bW0s  = (__bf16*)carve((size_t)GH * GF * 2);
  __bf16* bW0d  = (__bf16*)carve((size_t)GH * GF * 2);
  __bf16* bW1s  = (__bf16*)carve((size_t)GH * GH * 2);
  __bf16* bW1d  = (__bf16*)carve((size_t)GH * GH * 2);
  __bf16* bWih  = (__bf16*)carve((size_t)4 * GH * GH * 2);
  __bf16* bWhh  = (__bf16*)carve((size_t)4 * GH * GH * 2);
  __bf16* bWp   = (__bf16*)carve((size_t)GF * GH * 2);

  const int M = GG * GN;   // 128000 rows through both GCN layers

  // ---- zero accumulators (capture-safe async memsets) ----
  hipMemsetAsync(degO, 0, (size_t)GG * GN * 4, stream);
  hipMemsetAsync(degI, 0, (size_t)GG * GN * 4, stream);
  hipMemsetAsync(aggF, 0, (size_t)M * GF * 4, stream);
  hipMemsetAsync(aggB, 0, (size_t)M * GF * 4, stream);
  hipMemsetAsync(hst,  0, (size_t)BN * GH * 4, stream);
  hipMemsetAsync(cst,  0, (size_t)BN * GH * 4, stream);

  // ---- pre-convert weights to bf16 (constant per call) ----
  {
    struct { const float* s; __bf16* d; int n; } cv[7] = {
      {W0_s, bW0s, GH * GF}, {W0_d, bW0d, GH * GF},
      {W1_s, bW1s, GH * GH}, {W1_d, bW1d, GH * GH},
      {Wih,  bWih, 4 * GH * GH}, {Whh, bWhh, 4 * GH * GH},
      {Wp,   bWp,  GF * GH},
    };
    for (int i = 0; i < 7; ++i)
      k_cvt_bf16<<<(cv[i].n + 255) / 256, 256, 0, stream>>>(cv[i].s, cv[i].d, cv[i].n);
  }

  // ---- degree / normalization ----
  {
    int n = GG * GE;
    k_degrees<<<(n + 255) / 256, 256, 0, stream>>>(ei, ew, degO, degI);
    int m = 2 * GG * GN;   // degO & degI are adjacent 256B-aligned regions? no: separate launches
    k_invsqrt<<<((GG * GN) + 255) / 256, 256, 0, stream>>>(degO, GG * GN);
    k_invsqrt<<<((GG * GN) + 255) / 256, 256, 0, stream>>>(degI, GG * GN);
    (void)m;
    k_norm<<<(n + 255) / 256, 256, 0, stream>>>(ei, ew, degO, degI, nrm);
  }

  // ---- GCN layer 0: scatter (Fin=64) + WMMA projection -> hbuf ----
  {
    int tot = GG * GE * (GF / 4);
    k_scatter<<<(tot + 255) / 256, 256, 0, stream>>>(ei, nrm, x_seq, aggF, aggB, GF);
    k_gcn_gemm<GF><<<(M / 16) / 8, 256, 0, stream>>>(aggF, aggB, bW0s, bW0d,
                                                     b0_s, b0_d, hbuf, M);
  }

  // ---- GCN layer 1: re-zero, scatter (Fin=128) + WMMA projection -> hbuf ----
  hipMemsetAsync(aggF, 0, (size_t)M * GH * 4, stream);
  hipMemsetAsync(aggB, 0, (size_t)M * GH * 4, stream);
  {
    int tot = GG * GE * (GH / 4);
    k_scatter<<<(tot + 255) / 256, 256, 0, stream>>>(ei, nrm, hbuf, aggF, aggB, GH);
    k_gcn_gemm<GH><<<(M / 16) / 8, 256, 0, stream>>>(aggF, aggB, bW1s, bW1d,
                                                     b1_s, b1_d, hbuf, M);
  }

  // ---- LSTM over 8 steps (hbuf reinterpreted as [BN, 8, 128] row-major) ----
  {
    int un = BN * GH;
    for (int t = 0; t < GS; ++t) {
      k_lstm_gates<<<BN / 16, 256, 0, stream>>>(hbuf, hst, bWih, bWhh,
                                                bih, bhh, gates, t);
      k_lstm_update<<<(un + 255) / 256, 256, 0, stream>>>(gates, cst, hst);
    }
  }

  // ---- output projection ----
  k_proj<<<(BN / 16) / 8, 256, 0, stream>>>(hst, bWp, bp, out);
}
